// LSTMLanguageModel2_9552007266951
// MI455X (gfx1250) — matrix operand order repaired and checked by reading, not compiled, not run on hardware
//
#include <hip/hip_runtime.h>
#include <stdint.h>

// ---------------- problem constants ----------------
#define B_   32
#define S_   512
#define D_   128
#define H_   512
#define V_   10000
#define G4   2048           // 4*H
#define KC   2              // num cells
#define KDIM 640            // D + H
#define NTV  625            // V / 16
#define LSTM_WGS 64
#define OUT_ROWS 128        // rows of hs per output-GEMM workgroup
#define OUT_MT   8          // M tiles per workgroup

typedef __attribute__((ext_vector_type(16))) __bf16 bf16x16;
typedef __attribute__((ext_vector_type(8)))  float  floatx8;

union FragAB {
    bf16x16 v;
    uint4   q[2];
};

__device__ __forceinline__ uint16_t f2bf(float f) {
    union { float f; uint32_t u; } v; v.f = f;
    uint32_t u = v.u;
    return (uint16_t)((u + 0x7FFFu + ((u >> 16) & 1u)) >> 16);
}

__device__ __forceinline__ float sigmoidf_(float x) {
    return 1.0f / (1.0f + __expf(-x));
}

// ---- gfx1250 async global->LDS copy (ASYNCcnt-tracked, ISA 08_async_tensor) ----
__device__ __forceinline__ void async_ld_b128(uint32_t lds_off, const void* gaddr) {
    asm volatile("global_load_async_to_lds_b128 %0, %1, off"
                 :: "v"(lds_off), "v"(gaddr) : "memory");
}
__device__ __forceinline__ void wait_asynccnt0() {
    asm volatile("s_wait_asynccnt 0x0" ::: "memory");
}

// ---------------- grid-wide barrier (sense-reversal, graph-safe) ----------------
__device__ __forceinline__ void grid_barrier(unsigned* bar, unsigned nwg) {
    __syncthreads();
    if (threadIdx.x == 0) {
        __threadfence();
        unsigned gen  = __hip_atomic_load(&bar[1], __ATOMIC_RELAXED, __HIP_MEMORY_SCOPE_AGENT);
        unsigned prev = __hip_atomic_fetch_add(&bar[0], 1u, __ATOMIC_ACQ_REL, __HIP_MEMORY_SCOPE_AGENT);
        if (prev == nwg - 1u) {
            __hip_atomic_store(&bar[0], 0u, __ATOMIC_RELAXED, __HIP_MEMORY_SCOPE_AGENT);
            __hip_atomic_store(&bar[1], gen + 1u, __ATOMIC_RELEASE, __HIP_MEMORY_SCOPE_AGENT);
        } else {
            while (__hip_atomic_load(&bar[1], __ATOMIC_ACQUIRE, __HIP_MEMORY_SCOPE_AGENT) == gen) {
                __builtin_amdgcn_s_sleep(2);
            }
        }
        __threadfence();
    }
    __syncthreads();
}

// ---------------- kernel 1: weight prep (fp32 -> bf16, fuse Wih|Whh, sum biases) ----------------
__global__ void k_prep(const float* __restrict__ Wih, const float* __restrict__ Whh,
                       const float* __restrict__ bih, const float* __restrict__ bhh,
                       const float* __restrict__ Wout,
                       uint16_t* __restrict__ Wcat, float* __restrict__ bias,
                       uint16_t* __restrict__ Woutb) {
    const int NW = KC * G4 * KDIM;          // 2,621,440
    const int NB = KC * G4;                 // 4,096
    const int NO = V_ * H_;                 // 5,120,000
    int idx = blockIdx.x * blockDim.x + threadIdx.x;
    if (idx < NW) {
        int col  = idx % KDIM;
        int row  = (idx / KDIM) % G4;
        int cell = idx / (KDIM * G4);
        float v = (col < D_) ? Wih[(cell * G4 + row) * D_ + col]
                             : Whh[(cell * G4 + row) * H_ + (col - D_)];
        Wcat[idx] = f2bf(v);
    } else if (idx < NW + NB) {
        int j = idx - NW;
        bias[j] = bih[j] + bhh[j];
    } else if (idx < NW + NB + NO) {
        int j = idx - NW - NB;
        Woutb[j] = f2bf(Wout[j]);
    }
}

// ---------------- kernel 2: embedding gather -> time-major bf16 ----------------
__global__ void k_embed(const int* __restrict__ x, const float* __restrict__ emb,
                        uint16_t* __restrict__ xemb) {
    int idx = blockIdx.x * blockDim.x + threadIdx.x;   // S*B*D = 2,097,152
    if (idx >= S_ * B_ * D_) return;
    int d = idx & (D_ - 1);
    int b = (idx >> 7) & (B_ - 1);
    int t = idx >> 12;
    int id = x[b * S_ + t];
    xemb[idx] = f2bf(emb[(size_t)id * D_ + d]);
}

// ---------------- kernel 3: persistent LSTM recurrence (no LDS; L2-direct frags) ----------------
// hsx has S_+1 time rows: row 0 is h0 (zeros); phase C of step t writes row t+1.
__global__ void __launch_bounds__(256) k_lstm(
        const int* __restrict__ xids,          // (B,S)
        const uint16_t* __restrict__ xemb,     // (S,B,D) bf16
        const uint16_t* __restrict__ Wcat,     // (KC, G4, KDIM) bf16
        const float* __restrict__ bias,        // (KC, G4)
        float* __restrict__ c,                 // (B,H)
        float* __restrict__ gates,             // (KC, B, G4)
        uint16_t* hsx,                         // (S+1, B, H) bf16
        unsigned* __restrict__ bar) {
    const int tid  = threadIdx.x;
    const int lane = tid & 31;
    const int wave = tid >> 5;
    const int gw   = blockIdx.x * 8 + wave;       // 0..511 : one 16x16 tile per wave
    const int cell  = gw >> 8;                    // 0..1
    const int mtile = (gw >> 7) & 1;              // 0..1
    const int ntile = gw & 127;                   // 0..127

    const int lrow = lane & 15;
    const int lhi  = lane >> 4;
    const int arow = mtile * 16 + lrow;           // batch row for A fragments

    // ---- B tile (16 rows of Wcat) is invariant across all timesteps:
    //      hold all 20 K-fragments in registers for the whole recurrence ----
    FragAB fbv[KDIM / 32];
    {
        const int wrow = cell * G4 + ntile * 16 + lrow;
        const uint16_t* brow = Wcat + (size_t)wrow * KDIM + lhi * 8;
        #pragma unroll
        for (int j = 0; j < KDIM / 32; ++j) {
            fbv[j].q[0] = *reinterpret_cast<const uint4*>(brow + j * 32);
            fbv[j].q[1] = *reinterpret_cast<const uint4*>(brow + j * 32 + 16);
        }
    }

    for (int t = 0; t < S_; ++t) {
        // ---- phase B: batch-load all 20 A fragments (L2-resident), then 20 WMMAs ----
        const uint16_t* xrow = xemb + ((size_t)t * B_ + arow) * D_ + lhi * 8;
        const uint16_t* hrow = hsx  + ((size_t)t * B_ + arow) * H_ + lhi * 8;
        FragAB fa[KDIM / 32];
        #pragma unroll
        for (int j = 0; j < D_ / 32; ++j) {               // x_t part (K = 0..127)
            fa[j].q[0] = *reinterpret_cast<const uint4*>(xrow + j * 32);
            fa[j].q[1] = *reinterpret_cast<const uint4*>(xrow + j * 32 + 16);
        }
        #pragma unroll
        for (int j = 0; j < H_ / 32; ++j) {               // h_{t-1} part (K = 128..639)
            fa[D_ / 32 + j].q[0] = *reinterpret_cast<const uint4*>(hrow + j * 32);
            fa[D_ / 32 + j].q[1] = *reinterpret_cast<const uint4*>(hrow + j * 32 + 16);
        }
        floatx8 acc = {};
        #pragma unroll
        for (int j = 0; j < KDIM / 32; ++j) {
            acc = __builtin_amdgcn_wmma_f32_16x16x32_bf16(
                false, fa[j].v, false, fbv[j].v, (short)0, acc, false, false);
        }
        {
            const int col = ntile * 16 + lrow;
            const float bv = bias[cell * G4 + col];
            #pragma unroll
            for (int r = 0; r < 8; ++r) {
                int m = mtile * 16 + lhi * 8 + r;            // batch row 0..31
                gates[((size_t)cell * B_ + m) * G4 + col] = acc[r] + bv;
            }
        }

        grid_barrier(bar, LSTM_WGS);

        // ---- phase C: elementwise gate math + parity cell select ----
        {
            int idx = blockIdx.x * 256 + tid;                // exactly B*H = 16384 threads
            int b   = idx >> 9;
            int hh  = idx & (H_ - 1);
            float cprev = c[idx];
            float cn[KC], hn[KC];
            #pragma unroll
            for (int k = 0; k < KC; ++k) {
                const float* gp = gates + ((size_t)k * B_ + b) * G4;
                float gi = sigmoidf_(gp[hh]);
                float gf = sigmoidf_(gp[H_ + hh]);
                float gg = tanhf(gp[2 * H_ + hh]);
                float go = sigmoidf_(gp[3 * H_ + hh]);
                float cv = gf * cprev + gi * gg;
                cn[k] = cv;
                hn[k] = go * tanhf(cv);
            }
            int sel = xids[b * S_ + t] & 1;
            float cv = sel ? cn[1] : cn[0];
            float hv = sel ? hn[1] : hn[0];
            c[idx] = cv;
            hsx[((size_t)(t + 1) * B_ + b) * H_ + hh] = f2bf(hv);
        }

        grid_barrier(bar, LSTM_WGS);
    }
}

// ---------------- kernel 4: output projection (raw logits) ----------------
// 128 rows per WG; per N-tile: preload fb[16] once, then per M-tile batch-load
// fa[16] from LDS -> one wait -> 16 back-to-back WMMAs.
__global__ void __launch_bounds__(256) k_outgemm(
        const uint16_t* __restrict__ hs,       // (S*B, H) bf16
        const uint16_t* __restrict__ Woutb,    // (V, H) bf16
        const float* __restrict__ bout,        // (V)
        float* __restrict__ out) {             // (S*B, V)
    __shared__ __align__(16) uint16_t Hlds[OUT_ROWS][H_ + 8];   // 128 x 520 bf16 = 133,120 B

    const int tid  = threadIdx.x;
    const int lane = tid & 31;
    const int wave = tid >> 5;
    const int r0   = blockIdx.x * OUT_ROWS;
    const uint32_t HldsBase = (uint32_t)(uintptr_t)&Hlds[0][0];

    // async stage 128 x 512 bf16 (128 KB) global -> LDS
    for (int ci = tid; ci < OUT_ROWS * (H_ / 8); ci += 256) {    // 8192 chunks
        int row = ci >> 6, col8 = (ci & 63) * 8;
        async_ld_b128(HldsBase + row * (H_ + 8) * 2 + col8 * 2,
                      hs + (size_t)(r0 + row) * H_ + col8);
    }
    wait_asynccnt0();
    __syncthreads();

    const int lrow = lane & 15;
    const int lhi  = lane >> 4;

    for (int nt = wave; nt < NTV; nt += 8) {
        const int vrow = nt * 16 + lrow;
        const uint16_t* brow = Woutb + (size_t)vrow * H_ + lhi * 8;
        __builtin_prefetch(brow + (size_t)128 * H_, 0, 1);       // next nt for this wave

        FragAB fb[H_ / 32];
        #pragma unroll
        for (int j = 0; j < H_ / 32; ++j) {
            fb[j].q[0] = *reinterpret_cast<const uint4*>(brow + j * 32);
            fb[j].q[1] = *reinterpret_cast<const uint4*>(brow + j * 32 + 16);
        }

        const int v = nt * 16 + lrow;
        const float bv = bout[v];

        for (int mt = 0; mt < OUT_MT; ++mt) {
            FragAB fa[H_ / 32];
            #pragma unroll
            for (int j = 0; j < H_ / 32; ++j) {
                const uint16_t* ap = &Hlds[mt * 16 + lrow][j * 32 + lhi * 8];
                fa[j].q[0] = *reinterpret_cast<const uint4*>(ap);
                fa[j].q[1] = *reinterpret_cast<const uint4*>(ap + 16);
            }
            floatx8 acc = {};
            #pragma unroll
            for (int j = 0; j < H_ / 32; ++j) {
                acc = __builtin_amdgcn_wmma_f32_16x16x32_bf16(
                    false, fa[j].v, false, fb[j].v, (short)0, acc, false, false);
            }
            #pragma unroll
            for (int r = 0; r < 8; ++r) {
                int m = mt * 16 + lhi * 8 + r;
                out[(size_t)(r0 + m) * V_ + v] = acc[r] + bv;
            }
        }
    }
}

// ---------------- kernel 5: row-wise log-softmax (in place) ----------------
__global__ void __launch_bounds__(256) k_logsoftmax(float* __restrict__ out) {
    __shared__ float rowbuf[V_];
    __shared__ float red[256];
    const int tid = threadIdx.x;
    float* rowp = out + (size_t)blockIdx.x * V_;

    float lmax = -INFINITY;
    for (int i = tid; i < V_; i += 256) {
        float v = rowp[i];
        rowbuf[i] = v;
        lmax = fmaxf(lmax, v);
    }
    red[tid] = lmax;
    __syncthreads();
    for (int s = 128; s > 0; s >>= 1) {
        if (tid < s) red[tid] = fmaxf(red[tid], red[tid + s]);
        __syncthreads();
    }
    const float mx = red[0];
    __syncthreads();

    float lsum = 0.0f;
    for (int i = tid; i < V_; i += 256) lsum += __expf(rowbuf[i] - mx);
    red[tid] = lsum;
    __syncthreads();
    for (int s = 128; s > 0; s >>= 1) {
        if (tid < s) red[tid] += red[tid + s];
        __syncthreads();
    }
    const float lse = mx + logf(red[0]);
    __syncthreads();

    for (int i = tid; i < V_; i += 256) rowp[i] = rowbuf[i] - lse;
}

// ---------------- host launcher ----------------
extern "C" void kernel_launch(void* const* d_in, const int* in_sizes, int n_in,
                              void* d_out, int out_size, void* d_ws, size_t ws_size,
                              hipStream_t stream) {
    const int*   x    = (const int*)  d_in[0];
    const float* emb  = (const float*)d_in[1];
    const float* Wih  = (const float*)d_in[2];
    const float* Whh  = (const float*)d_in[3];
    const float* bih  = (const float*)d_in[4];
    const float* bhh  = (const float*)d_in[5];
    const float* Wout = (const float*)d_in[6];
    const float* bout = (const float*)d_in[7];
    float* out = (float*)d_out;

    char* ws = (char*)d_ws;
    size_t off = 0;
    auto alloc = [&](size_t bytes) { size_t o = off; off = (off + bytes + 255) & ~(size_t)255; return o; };
    uint16_t* Wcat  = (uint16_t*)(ws + alloc((size_t)KC * G4 * KDIM * 2));   // 5.24 MB
    float*    bias  = (float*)   (ws + alloc((size_t)KC * G4 * 4));
    uint16_t* Woutb = (uint16_t*)(ws + alloc((size_t)V_ * H_ * 2));          // 10.2 MB
    uint16_t* xemb  = (uint16_t*)(ws + alloc((size_t)S_ * B_ * D_ * 2));     // 4.2 MB
    float*    cbuf  = (float*)   (ws + alloc((size_t)B_ * H_ * 4));
    float*    gates = (float*)   (ws + alloc((size_t)KC * B_ * G4 * 4));
    uint16_t* hsx   = (uint16_t*)(ws + alloc((size_t)(S_ + 1) * B_ * H_ * 2)); // 16.8 MB
    unsigned* bar   = (unsigned*)(ws + alloc(256));
    (void)ws_size; (void)n_in; (void)in_sizes; (void)out_size;

    hipMemsetAsync(cbuf, 0, (size_t)B_ * H_ * 4, stream);
    hipMemsetAsync(hsx,  0, (size_t)B_ * H_ * 2, stream);    // h0 = 0
    hipMemsetAsync(bar,  0, 256, stream);

    {
        int total = KC * G4 * KDIM + KC * G4 + V_ * H_;
        k_prep<<<(total + 255) / 256, 256, 0, stream>>>(Wih, Whh, bih, bhh, Wout,
                                                        Wcat, bias, Woutb);
    }
    {
        int total = S_ * B_ * D_;
        k_embed<<<(total + 255) / 256, 256, 0, stream>>>(x, emb, xemb);
    }
    k_lstm<<<LSTM_WGS, 256, 0, stream>>>(x, xemb, Wcat, bias, cbuf, gates, hsx, bar);
    k_outgemm<<<(S_ * B_) / OUT_ROWS, 256, 0, stream>>>(hsx + (size_t)B_ * H_,
                                                        Woutb, bout, out);
    k_logsoftmax<<<S_ * B_, 256, 0, stream>>>(out);
}
